// TensorBase_67250597921305
// MI455X (gfx1250) — matrix operand order repaired and checked by reading, not compiled, not run on hardware
//
#include <hip/hip_runtime.h>
#include <math.h>

typedef __attribute__((ext_vector_type(16))) _Float16 v16h;
typedef __attribute__((ext_vector_type(8)))  float    v8f;

#define NRAYS   2048
#define NSAMP   256
#define NP      (NRAYS*NSAMP)
#define GRIDN   128
#define GRID2   (GRIDN*GRIDN)
#define GRID3   (GRIDN*GRIDN*GRIDN)
#define APPD    27
#define FEATCN  128
#define KIN     150
#define KP      160              // K for layer 1 padded to 5 chunks of 32
#define TNEAR   2.0f
#define TFAR    6.0f
#define DELTA   ((TFAR-TNEAR)/(float)(NSAMP-1))
#define DSHIFT  (-10.0f)
#define DSCALE  25.0f

// workspace layout (bytes)
#define WS_W1   0                                  // [128][160] f16 (n-major, transposed)
#define WS_W2   (WS_W1 + FEATCN*KP*2)              // [128][128] f16 (n-major, transposed)
#define WS_WGT  (WS_W2 + FEATCN*FEATCN*2)          // [NP] f32 per-point weights
#define WS_ACC  (WS_WGT + NP*4)                    // [NRAYS] f32 acc_map

union HU { v16h h; unsigned u[8]; };
union PK8 { int4 i; _Float16 h[8]; };

// ---------------- weight conversion: f32 -> transposed padded f16 ----------------
__global__ void k_convert(const float* __restrict__ w1, const float* __restrict__ w2,
                          char* __restrict__ ws) {
  int i = blockIdx.x * blockDim.x + threadIdx.x;
  _Float16* w1t = (_Float16*)(ws + WS_W1);
  _Float16* w2t = (_Float16*)(ws + WS_W2);
  if (i < FEATCN * KP) {
    int n = i / KP, k = i % KP;
    float v = (k < KIN) ? w1[k * FEATCN + n] : 0.0f;
    w1t[i] = (_Float16)v;
  }
  int j = i - FEATCN * KP;
  if (j >= 0 && j < FEATCN * FEATCN) {
    int n = j / FEATCN, k = j % FEATCN;
    w2t[j] = (_Float16)w2[k * FEATCN + n];
  }
}

__global__ void k_zero(float* __restrict__ out) {
  int i = blockIdx.x * blockDim.x + threadIdx.x;
  if (i < NRAYS * 3) out[i] = 0.0f;
}

// ---------------- per-ray sigma/alpha/transmittance/weights ----------------
__global__ __launch_bounds__(NSAMP) void k_weights(const float* __restrict__ rays_o,
                                                   const float* __restrict__ rays_d,
                                                   const float* __restrict__ dvol,
                                                   char* __restrict__ ws) {
  __shared__ float sh[NSAMP];
  int r = blockIdx.x;
  int s = threadIdx.x;

  float dx = rays_d[r*3+0], dy = rays_d[r*3+1], dz = rays_d[r*3+2];
  float il = rsqrtf(dx*dx + dy*dy + dz*dz);
  dx *= il; dy *= il; dz *= il;
  float zv = TNEAR + (TFAR - TNEAR) * (float)s / (float)(NSAMP - 1);
  float px = rays_o[r*3+0] + dx*zv;
  float py = rays_o[r*3+1] + dy*zv;
  float pz = rays_o[r*3+2] + dz*zv;

  float cx = fminf(fmaxf((px + 1.5f) * (2.0f/3.0f) - 1.0f, -1.0f), 1.0f);
  float cy = fminf(fmaxf((py + 1.5f) * (2.0f/3.0f) - 1.0f, -1.0f), 1.0f);
  float cz = fminf(fmaxf((pz + 1.5f) * (2.0f/3.0f) - 1.0f, -1.0f), 1.0f);
  float gx = (cx + 1.0f) * 0.5f * (float)(GRIDN - 1);
  float gy = (cy + 1.0f) * 0.5f * (float)(GRIDN - 1);
  float gz = (cz + 1.0f) * 0.5f * (float)(GRIDN - 1);
  int x0 = (int)floorf(gx); x0 = x0 < 0 ? 0 : (x0 > GRIDN-2 ? GRIDN-2 : x0);
  int y0 = (int)floorf(gy); y0 = y0 < 0 ? 0 : (y0 > GRIDN-2 ? GRIDN-2 : y0);
  int z0 = (int)floorf(gz); z0 = z0 < 0 ? 0 : (z0 > GRIDN-2 ? GRIDN-2 : z0);
  float fx = gx - (float)x0, fy = gy - (float)y0, fz = gz - (float)z0;

  const float* v0 = dvol + (size_t)z0*GRID2 + (size_t)y0*GRIDN + (size_t)x0;
  const float* v1 = v0 + GRID2;
  float c00 = v0[0]      * (1.0f-fx) + v0[1]       * fx;
  float c01 = v0[GRIDN]  * (1.0f-fx) + v0[GRIDN+1] * fx;
  float c10 = v1[0]      * (1.0f-fx) + v1[1]       * fx;
  float c11 = v1[GRIDN]  * (1.0f-fx) + v1[GRIDN+1] * fx;
  float sf = (c00*(1.0f-fy) + c01*fy) * (1.0f-fz) + (c10*(1.0f-fy) + c11*fy) * fz + DSHIFT;
  float sigma = (sf > 20.0f) ? sf : __logf(1.0f + __expf(sf));
  float alpha = 1.0f - __expf(-sigma * DELTA * DSCALE);

  // exclusive prefix product of (1 - alpha + 1e-10) -> transmittance
  float tv = 1.0f - alpha + 1e-10f;
  sh[s] = tv;
  __syncthreads();
  float incl = tv;
  for (int off = 1; off < NSAMP; off <<= 1) {
    float o = (s >= off) ? sh[s - off] : 1.0f;
    __syncthreads();
    incl *= o;
    sh[s] = incl;
    __syncthreads();
  }
  float Texc = (s == 0) ? 1.0f : sh[s - 1];
  float wgt = alpha * Texc;
  ((float*)(ws + WS_WGT))[r * NSAMP + s] = wgt;

  __syncthreads();
  sh[s] = wgt;
  __syncthreads();
  for (int off = NSAMP/2; off > 0; off >>= 1) {
    if (s < off) sh[s] += sh[s + off];
    __syncthreads();
  }
  if (s == 0) ((float*)(ws + WS_ACC))[r] = sh[0];
}

// ---------------- fused feature + 3-layer MLP (WMMA) + weighted accumulate ----------------
__global__ __launch_bounds__(256) void k_mlp(const float* __restrict__ rays_o,
                                             const float* __restrict__ rays_d,
                                             const float* __restrict__ avol,
                                             const float* __restrict__ b1,
                                             const float* __restrict__ b2,
                                             const float* __restrict__ w3,
                                             const float* __restrict__ b3,
                                             char* __restrict__ ws,
                                             float* __restrict__ out) {
  __shared__ __align__(16) _Float16 sIn[128 * KP];        // 40 KB: features / h1 / h2
  __shared__ __align__(16) _Float16 sW1[FEATCN * KP];     // 40 KB: w1^T [n][160]
  __shared__ __align__(16) _Float16 sW2[FEATCN * FEATCN]; // 32 KB: w2^T [n][128]
  __shared__ __align__(16) float    sW3[FEATCN * 3];      // 1.5 KB
  __shared__ float sB1[FEATCN];
  __shared__ float sB2[FEATCN];
  __shared__ float sB3[4];
  __shared__ float sred[3][128];
  const int t  = threadIdx.x;
  const int p0 = blockIdx.x * 128;     // 128 consecutive points (half of one ray)

  // ---- stage 0: cooperative weight staging into LDS (b128 loads + ds stores) ----
  {
    const int4* s1 = (const int4*)(ws + WS_W1);
    const int4* s2 = (const int4*)(ws + WS_W2);
    int4* d1 = (int4*)sW1;
    int4* d2 = (int4*)sW2;
    #pragma unroll
    for (int i = 0; i < (FEATCN*KP*2)/16/256; ++i)        // 10 per thread
      d1[t + i*256] = s1[t + i*256];
    #pragma unroll
    for (int i = 0; i < (FEATCN*FEATCN*2)/16/256; ++i)    // 8 per thread
      d2[t + i*256] = s2[t + i*256];
    for (int i = t; i < FEATCN*3; i += 256) sW3[i] = w3[i];
    if (t < FEATCN) { sB1[t] = b1[t]; sB2[t] = b2[t]; }
    if (t < 3) sB3[t] = b3[t];
  }

  // ---- stage 1: per-point 150-dim feature row -> LDS, 2 threads per point ----
  {
    int pt   = t >> 1;           // 0..127
    int half = t & 1;            // channel parity
    int p    = p0 + pt;
    int ray  = p >> 8;
    int s    = p & (NSAMP - 1);
    float dx = rays_d[ray*3+0], dy = rays_d[ray*3+1], dz = rays_d[ray*3+2];
    float il = rsqrtf(dx*dx + dy*dy + dz*dz);
    dx *= il; dy *= il; dz *= il;
    float zv = TNEAR + (TFAR - TNEAR) * (float)s / (float)(NSAMP - 1);
    float px = rays_o[ray*3+0] + dx*zv;
    float py = rays_o[ray*3+1] + dy*zv;
    float pz = rays_o[ray*3+2] + dz*zv;

    float cx = fminf(fmaxf((px + 1.5f) * (2.0f/3.0f) - 1.0f, -1.0f), 1.0f);
    float cy = fminf(fmaxf((py + 1.5f) * (2.0f/3.0f) - 1.0f, -1.0f), 1.0f);
    float cz = fminf(fmaxf((pz + 1.5f) * (2.0f/3.0f) - 1.0f, -1.0f), 1.0f);
    float gx = (cx + 1.0f) * 0.5f * (float)(GRIDN - 1);
    float gy = (cy + 1.0f) * 0.5f * (float)(GRIDN - 1);
    float gz = (cz + 1.0f) * 0.5f * (float)(GRIDN - 1);
    int x0 = (int)floorf(gx); x0 = x0 < 0 ? 0 : (x0 > GRIDN-2 ? GRIDN-2 : x0);
    int y0 = (int)floorf(gy); y0 = y0 < 0 ? 0 : (y0 > GRIDN-2 ? GRIDN-2 : y0);
    int z0 = (int)floorf(gz); z0 = z0 < 0 ? 0 : (z0 > GRIDN-2 ? GRIDN-2 : z0);
    float fx = gx - (float)x0, fy = gy - (float)y0, fz = gz - (float)z0;
    float w000 = (1.0f-fz)*(1.0f-fy)*(1.0f-fx);
    float w001 = (1.0f-fz)*(1.0f-fy)*fx;
    float w010 = (1.0f-fz)*fy*(1.0f-fx);
    float w011 = (1.0f-fz)*fy*fx;
    float w100 = fz*(1.0f-fy)*(1.0f-fx);
    float w101 = fz*(1.0f-fy)*fx;
    float w110 = fz*fy*(1.0f-fx);
    float w111 = fz*fy*fx;
    size_t base = (size_t)z0*GRID2 + (size_t)y0*GRIDN + (size_t)x0;

    _Float16* row = &sIn[pt * KP];
    // channels with parity `half` (even thread: 14, odd thread: 13)
    #pragma unroll
    for (int cc = 0; cc < 14; ++cc) {
      int c = 2*cc + half;
      if (c < APPD) {
        const float* vc = avol + (size_t)c * GRID3 + base;
        const float* vz = vc + GRID2;
        float f = vc[0]*w000 + vc[1]*w001 + vc[GRIDN]*w010 + vc[GRIDN+1]*w011
                + vz[0]*w100 + vz[1]*w101 + vz[GRIDN]*w110 + vz[GRIDN+1]*w111;
        row[c] = (_Float16)f;
        row[30 + 2*c] = (_Float16)__sinf(f);
        row[31 + 2*c] = (_Float16)__sinf(2.0f * f);
        row[84 + 2*c] = (_Float16)__cosf(f);
        row[85 + 2*c] = (_Float16)__cosf(2.0f * f);
      }
    }
    if (half == 0) {
      row[27] = (_Float16)dx; row[28] = (_Float16)dy; row[29] = (_Float16)dz;
      float vdv[3] = {dx, dy, dz};
      #pragma unroll
      for (int c = 0; c < 3; ++c) {
        float f = vdv[c];
        row[138 + 2*c] = (_Float16)__sinf(f);
        row[139 + 2*c] = (_Float16)__sinf(2.0f * f);
        row[144 + 2*c] = (_Float16)__cosf(f);
        row[145 + 2*c] = (_Float16)__cosf(2.0f * f);
      }
      #pragma unroll
      for (int k = KIN; k < KP; ++k) row[k] = (_Float16)0.0f;
    }
  }
  __syncthreads();

  // ---- stage 2: WMMA layers, one wave per 16-row stripe, 8 N-tiles each ----
  const int lane = t & 31;
  const int wv   = t >> 5;
  const int m0   = wv * 16;
  const int hi   = lane >> 4;
  const int ln   = lane & 15;

  v8f acc[8];

  // ----- layer 1: [128x160] @ [160x128], bias folded into C -----
  #pragma unroll
  for (int nt = 0; nt < 8; ++nt) {
    float bb = sB1[nt * 16 + ln];
    #pragma unroll
    for (int j = 0; j < 8; ++j) acc[nt][j] = bb;
  }
  #pragma unroll
  for (int kc = 0; kc < 5; ++kc) {
    HU a;
    #pragma unroll
    for (int j = 0; j < 8; ++j) {
      int k = kc * 32 + ((j < 4) ? (2*j + 8*hi) : (16 + 2*(j-4) + 8*hi));
      a.u[j] = *(const unsigned*)&sIn[(m0 + ln) * KP + k];
    }
    #pragma unroll
    for (int nt = 0; nt < 8; ++nt) {
      HU b;
      #pragma unroll
      for (int j = 0; j < 8; ++j) {
        int k = kc * 32 + ((j < 4) ? (2*j + 8*hi) : (16 + 2*(j-4) + 8*hi));
        b.u[j] = *(const unsigned*)&sW1[(nt*16 + ln) * KP + k];
      }
      acc[nt] = __builtin_amdgcn_wmma_f32_16x16x32_f16(
          false, a.h, false, b.h, (short)0, acc[nt], false, false);
    }
  }
  // relu, restripe h1 into LDS (each wave touches only its own 16 rows)
  #pragma unroll
  for (int nt = 0; nt < 8; ++nt)
    #pragma unroll
    for (int v = 0; v < 8; ++v)
      sIn[(m0 + v + 8*hi) * KP + nt*16 + ln] = (_Float16)fmaxf(acc[nt][v], 0.0f);
  __syncthreads();

  // ----- layer 2: [128x128] @ [128x128] -----
  #pragma unroll
  for (int nt = 0; nt < 8; ++nt) {
    float bb = sB2[nt * 16 + ln];
    #pragma unroll
    for (int j = 0; j < 8; ++j) acc[nt][j] = bb;
  }
  #pragma unroll
  for (int kc = 0; kc < 4; ++kc) {
    HU a;
    #pragma unroll
    for (int j = 0; j < 8; ++j) {
      int k = kc * 32 + ((j < 4) ? (2*j + 8*hi) : (16 + 2*(j-4) + 8*hi));
      a.u[j] = *(const unsigned*)&sIn[(m0 + ln) * KP + k];
    }
    #pragma unroll
    for (int nt = 0; nt < 8; ++nt) {
      HU b;
      #pragma unroll
      for (int j = 0; j < 8; ++j) {
        int k = kc * 32 + ((j < 4) ? (2*j + 8*hi) : (16 + 2*(j-4) + 8*hi));
        b.u[j] = *(const unsigned*)&sW2[(nt*16 + ln) * FEATCN + k];
      }
      acc[nt] = __builtin_amdgcn_wmma_f32_16x16x32_f16(
          false, a.h, false, b.h, (short)0, acc[nt], false, false);
    }
  }
  #pragma unroll
  for (int nt = 0; nt < 8; ++nt)
    #pragma unroll
    for (int v = 0; v < 8; ++v)
      sIn[(m0 + v + 8*hi) * KP + nt*16 + ln] = (_Float16)fmaxf(acc[nt][v], 0.0f);
  __syncthreads();

  // ---- stage 3: layer 3 (128->3) + sigmoid, then block reduction + 3 atomics ----
  if (t < 128) {
    int p = p0 + t;
    float a0 = sB3[0], a1 = sB3[1], a2 = sB3[2];
    const _Float16* row = &sIn[t * KP];
    #pragma unroll
    for (int kb = 0; kb < FEATCN; kb += 8) {
      PK8 pk;
      pk.i = *(const int4*)&row[kb];
      #pragma unroll
      for (int j = 0; j < 8; ++j) {
        float h = (float)pk.h[j];
        a0 += h * sW3[(kb + j)*3 + 0];
        a1 += h * sW3[(kb + j)*3 + 1];
        a2 += h * sW3[(kb + j)*3 + 2];
      }
    }
    float wgt = ((const float*)(ws + WS_WGT))[p];
    sred[0][t] = wgt / (1.0f + __expf(-a0));
    sred[1][t] = wgt / (1.0f + __expf(-a1));
    sred[2][t] = wgt / (1.0f + __expf(-a2));
  }
  __syncthreads();
  #pragma unroll
  for (int off = 64; off > 0; off >>= 1) {
    if (t < off) {
      sred[0][t] += sred[0][t + off];
      sred[1][t] += sred[1][t + off];
      sred[2][t] += sred[2][t + off];
    }
    __syncthreads();
  }
  if (t == 0) {
    int ray = p0 >> 8;
    atomicAdd(&out[ray*3 + 0], sred[0][0]);
    atomicAdd(&out[ray*3 + 1], sred[1][0]);
    atomicAdd(&out[ray*3 + 2], sred[2][0]);
  }
}

__global__ void k_finalize(float* __restrict__ out, const char* __restrict__ ws) {
  int r = blockIdx.x * blockDim.x + threadIdx.x;
  if (r < NRAYS) {
    float bg = 1.0f - ((const float*)(ws + WS_ACC))[r];
    #pragma unroll
    for (int c = 0; c < 3; ++c) {
      float v = out[r*3 + c] + bg;
      out[r*3 + c] = fminf(fmaxf(v, 0.0f), 1.0f);
    }
  }
}

extern "C" void kernel_launch(void* const* d_in, const int* in_sizes, int n_in,
                              void* d_out, int out_size, void* d_ws, size_t ws_size,
                              hipStream_t stream) {
  (void)in_sizes; (void)n_in; (void)out_size; (void)ws_size;
  const float* rays_o = (const float*)d_in[0];
  const float* rays_d = (const float*)d_in[1];
  const float* dvol   = (const float*)d_in[2];
  const float* avol   = (const float*)d_in[3];
  const float* w1     = (const float*)d_in[4];
  const float* b1     = (const float*)d_in[5];
  const float* w2     = (const float*)d_in[6];
  const float* b2     = (const float*)d_in[7];
  const float* w3     = (const float*)d_in[8];
  const float* b3     = (const float*)d_in[9];
  float* out = (float*)d_out;
  char*  ws  = (char*)d_ws;

  int convN = FEATCN*KP + FEATCN*FEATCN;
  k_convert<<<(convN + 255)/256, 256, 0, stream>>>(w1, w2, ws);
  k_zero<<<(NRAYS*3 + 255)/256, 256, 0, stream>>>(out);
  k_weights<<<NRAYS, NSAMP, 0, stream>>>(rays_o, rays_d, dvol, ws);
  k_mlp<<<NP/128, 256, 0, stream>>>(rays_o, rays_d, avol, b1, b2, w3, b3, ws, out);
  k_finalize<<<(NRAYS + 255)/256, 256, 0, stream>>>(out, ws);
}
